// point_involution_v1_23278722744989
// MI455X (gfx1250) — matrix operand order repaired and checked by reading, not compiled, not run on hardware
//
#include <hip/hip_runtime.h>
#include <hip/hip_bf16.h>

// ---------------------------------------------------------------------------
// Point involution for MI455X (gfx1250, wave32, WMMA), compile-only tuning.
//
//   h    = s_feats @ w1 + b1            (bf16 WMMA, computed twice, never stored)
//   act  = LeakyReLU(BN(h))             (stats via per-block partials + finalize)
//   attn = act @ w2 + b2                (bf16 WMMA, tile lives in LDS only)
//   out[m,c] = sum_h s_feats[ind[m,h], c] * attn[m, h*16 + (c>>3)]
//
// WMMA lane layouts (CDNA5 ISA 7.12.2, 16-bit, wave32):
//   A (16x32, MxK): lane l holds row m = l&15.
//       l<16 : elems 0..7 = K(base+0..7),  elems 8..15 = K(base+16..23)
//       l>=16: elems 0..7 = K(base+8..15), elems 8..15 = K(base+24..31)
//   B (32x16, KxN): lane l holds col n = l&15.
//       l<16 : elems 0..7 = K(base+0..7), 8..15 = K(base+16..23); l>=16: K(base+8..15), K(base+24..31)
//   C/D (16x16 f32): lane l holds col n = l&15, VGPR i = row ((l<16?0:8)+i).
// Weights pre-packed into B layout -> contiguous 32B loads per lane per tile.
// ---------------------------------------------------------------------------

typedef __attribute__((ext_vector_type(16))) __bf16 v16bf;
typedef __attribute__((ext_vector_type(8)))  float  v8f;

#define C_DIM 128
#define H_NB  32
#define NEG_SLOPE 0.1f
#define BN_EPS 1e-5f
#define HT_STRIDE 132   // padded LDS row stride (floats) for the h tile

// Pack W (K=128 x N f32, row-major) into bf16 B-matrix lane layout:
// Wp[ntile][kchunk][lane][elem], ntiles = N/16, kchunks = 4.
__global__ void pack_w_bf16(const float* __restrict__ W,
                            __bf16* __restrict__ Wp,
                            int N, int ntiles) {
    int idx = blockIdx.x * blockDim.x + threadIdx.x;
    int total = ntiles * 4 * 32 * 16;
    if (idx >= total) return;
    int j    = idx & 15;
    int lane = (idx >> 4) & 31;
    int kc   = (idx >> 9) & 3;
    int nt   = idx >> 11;
    int n = (lane & 15) + 16 * nt;
    int k = kc * 32 + ((lane < 16) ? 0 : 8) + ((j < 8) ? j : 8 + j);
    Wp[idx] = (__bf16)W[k * N + n];
}

// Build the WMMA A tile (4 chunks of 16x32 bf16) for 16 rows starting at m0,
// reading row-major f32 data with leading dimension C_DIM.
__device__ __forceinline__ void load_A_from_global(const float* __restrict__ base,
                                                   int lane, v16bf A[4]) {
    const int koff = (lane < 16) ? 0 : 8;
    const float* p = base;  // caller positions at row
#pragma unroll
    for (int kc = 0; kc < 4; ++kc) {
        const int b = kc * 32 + koff;
#pragma unroll
        for (int j = 0; j < 8; ++j) {
            A[kc][j]     = (__bf16)p[b + j];
            A[kc][8 + j] = (__bf16)p[b + 16 + j];
        }
    }
}

// ---------------------------------------------------------------------------
// Pass 1: h-tile via WMMA, per-block per-channel partial sum / sumsq.
// partial layout: p[c * nblocks + blk]  (coalesced for the finalize kernel).
// ---------------------------------------------------------------------------
__global__ void __launch_bounds__(32)
gemm1_stats_kernel(const float* __restrict__ sfeat,
                   const __bf16* __restrict__ w1p,
                   const float* __restrict__ b1,
                   float* __restrict__ psum,
                   float* __restrict__ psq) {
    const int lane = threadIdx.x;
    const int m0   = blockIdx.x * 16;
    const int nblocks = gridDim.x;

    v16bf A[4];
    load_A_from_global(sfeat + (size_t)(m0 + (lane & 15)) * C_DIM, lane, A);

#pragma unroll
    for (int nt = 0; nt < 8; ++nt) {
        const int n = nt * 16 + (lane & 15);
        const float bias = b1[n];
        v8f acc = {0.f, 0.f, 0.f, 0.f, 0.f, 0.f, 0.f, 0.f};
#pragma unroll
        for (int kc = 0; kc < 4; ++kc) {
            v16bf B = *(const v16bf*)(w1p + ((size_t)(nt * 4 + kc) * 32 + lane) * 16);
            acc = __builtin_amdgcn_wmma_f32_16x16x32_bf16(
                false, A[kc], false, B, (short)0, acc, false, false);
        }
        float s = 0.f, q = 0.f;
#pragma unroll
        for (int i = 0; i < 8; ++i) {
            float v = acc[i] + bias;
            s += v;
            q += v * v;
        }
        // lanes l and l^16 hold the same channel n (different row halves)
        s += __shfl_xor(s, 16, 32);
        q += __shfl_xor(q, 16, 32);
        if (lane < 16) {
            psum[(size_t)n * nblocks + blockIdx.x] = s;
            psq[(size_t)n * nblocks + blockIdx.x]  = q;
        }
    }
}

// Finalize BN: one block per channel, coalesced partial reads -> scale/shift.
__global__ void __launch_bounds__(256)
bn_finalize_kernel(const float* __restrict__ psum,
                   const float* __restrict__ psq,
                   const float* __restrict__ gamma,
                   const float* __restrict__ beta,
                   float* __restrict__ scale,
                   float* __restrict__ shift,
                   int nblocks, int M) {
    __shared__ float ssum[256];
    __shared__ float ssq[256];
    const int c = blockIdx.x;
    const int t = threadIdx.x;
    float s = 0.f, q = 0.f;
    for (int b = t; b < nblocks; b += 256) {
        s += psum[(size_t)c * nblocks + b];
        q += psq[(size_t)c * nblocks + b];
    }
    ssum[t] = s; ssq[t] = q;
    __syncthreads();
    for (int o = 128; o > 0; o >>= 1) {
        if (t < o) { ssum[t] += ssum[t + o]; ssq[t] += ssq[t + o]; }
        __syncthreads();
    }
    if (t == 0) {
        float inv_m = 1.0f / (float)M;
        float mean = ssum[0] * inv_m;
        float var  = ssq[0] * inv_m - mean * mean;
        float sc = gamma[c] * rsqrtf(var + BN_EPS);
        scale[c] = sc;
        shift[c] = beta[c] - mean * sc;
    }
}

// ---------------------------------------------------------------------------
// Pass 2 (mega-fused): recompute h tile (WMMA) -> LDS transpose -> BN+LeakyReLU
// -> GEMM2 (WMMA, attn tile in LDS) -> neighbor gather / weighted reduce.
// One wave per 16 query rows. LDS: 16*132*4 + 16*512*4 = 41216 B.
// ---------------------------------------------------------------------------
__global__ void __launch_bounds__(32)
fused_kernel(const float* __restrict__ sfeat,
             const __bf16* __restrict__ w1p,
             const float* __restrict__ b1,
             const float* __restrict__ scale,
             const float* __restrict__ shift,
             const __bf16* __restrict__ w2p,
             const float* __restrict__ b2,
             const int* __restrict__ inds,
             float* __restrict__ out) {
    __shared__ float h_lds[16 * HT_STRIDE];
    __shared__ float attn_s[16 * 512];

    const int lane = threadIdx.x;
    const int m0   = blockIdx.x * 16;
    const int koff = (lane < 16) ? 0 : 8;
    const int r0   = (lane < 16) ? 0 : 8;

    // --- phase 1: h tile = s_feats @ w1 + b1 (WMMA), D layout -> LDS ---
    {
        v16bf A[4];
        load_A_from_global(sfeat + (size_t)(m0 + (lane & 15)) * C_DIM, lane, A);
#pragma unroll
        for (int nt = 0; nt < 8; ++nt) {
            const int n = nt * 16 + (lane & 15);
            const float bias = b1[n];
            v8f acc = {0.f, 0.f, 0.f, 0.f, 0.f, 0.f, 0.f, 0.f};
#pragma unroll
            for (int kc = 0; kc < 4; ++kc) {
                v16bf B = *(const v16bf*)(w1p + ((size_t)(nt * 4 + kc) * 32 + lane) * 16);
                acc = __builtin_amdgcn_wmma_f32_16x16x32_bf16(
                    false, A[kc], false, B, (short)0, acc, false, false);
            }
#pragma unroll
            for (int i = 0; i < 8; ++i)
                h_lds[(r0 + i) * HT_STRIDE + n] = acc[i] + bias;
        }
    }
    __syncthreads();

    // --- phase 2: act = LeakyReLU(BN(h)) back into WMMA A layout ---
    v16bf A2[4];
    {
        const float* hp = &h_lds[(lane & 15) * HT_STRIDE];
#pragma unroll
        for (int kc = 0; kc < 4; ++kc) {
            const int b = kc * 32 + koff;
#pragma unroll
            for (int j = 0; j < 8; ++j) {
                int k0 = b + j;
                int k1 = b + 16 + j;
                float v0 = hp[k0] * scale[k0] + shift[k0];
                float v1 = hp[k1] * scale[k1] + shift[k1];
                v0 = (v0 >= 0.f) ? v0 : NEG_SLOPE * v0;
                v1 = (v1 >= 0.f) ? v1 : NEG_SLOPE * v1;
                A2[kc][j]     = (__bf16)v0;
                A2[kc][8 + j] = (__bf16)v1;
            }
        }
    }

    // --- phase 3: attn tile = act @ w2 + b2 (32 N-tiles x 4 K-steps) ---
#pragma unroll 4
    for (int nt = 0; nt < 32; ++nt) {
        const int n = nt * 16 + (lane & 15);
        const float bias = b2[n];
        v8f acc = {0.f, 0.f, 0.f, 0.f, 0.f, 0.f, 0.f, 0.f};
#pragma unroll
        for (int kc = 0; kc < 4; ++kc) {
            v16bf B = *(const v16bf*)(w2p + ((size_t)(nt * 4 + kc) * 32 + lane) * 16);
            acc = __builtin_amdgcn_wmma_f32_16x16x32_bf16(
                false, A2[kc], false, B, (short)0, acc, false, false);
        }
#pragma unroll
        for (int i = 0; i < 8; ++i)
            attn_s[(r0 + i) * 512 + n] = acc[i] + bias;
    }
    __syncthreads();

    // --- phase 4: gather + weighted reduce ---
    // Lane owns channels c0..c0+3 (c0 = 4*lane) => one attn scalar per
    // neighbor: cpg = c0>>3 = lane>>1.  Per neighbor the wave reads a
    // contiguous 512B row of s_feats (L2-resident table).
    const int c0  = lane * 4;
    const int cpg = lane >> 1;
    for (int r = 0; r < 16; ++r) {
        const int m = m0 + r;
        const int* ip = inds + (size_t)m * H_NB;
        float a0 = 0.f, a1 = 0.f, a2 = 0.f, a3 = 0.f;
#pragma unroll 4
        for (int h = 0; h < H_NB; ++h) {
            const int idx = ip[h];
            const float4 nb = *(const float4*)(sfeat + (size_t)idx * C_DIM + c0);
            const float a = attn_s[r * 512 + h * 16 + cpg];
            a0 += nb.x * a;
            a1 += nb.y * a;
            a2 += nb.z * a;
            a3 += nb.w * a;
        }
        float4 o; o.x = a0; o.y = a1; o.z = a2; o.w = a3;
        *(float4*)(out + (size_t)m * C_DIM + c0) = o;
    }
}

extern "C" void kernel_launch(void* const* d_in, const int* in_sizes, int n_in,
                              void* d_out, int out_size, void* d_ws, size_t ws_size,
                              hipStream_t stream) {
    // Inputs (setup_inputs order):
    // 0 q_pts (unused), 1 s_pts (unused), 2 s_feats (N,128) f32,
    // 3 neighb_inds (M,32) i32, 4 w1 (128,128), 5 b1 (128),
    // 6 gamma1 (128), 7 beta1 (128), 8 w2 (128,512), 9 b2 (512)
    const float* s_feats = (const float*)d_in[2];
    const int*   inds    = (const int*)d_in[3];
    const float* w1      = (const float*)d_in[4];
    const float* b1      = (const float*)d_in[5];
    const float* gamma1  = (const float*)d_in[6];
    const float* beta1   = (const float*)d_in[7];
    const float* w2      = (const float*)d_in[8];
    const float* b2      = (const float*)d_in[9];
    float* out = (float*)d_out;

    const int M = in_sizes[3] / H_NB;   // 50000 (divisible by 16)
    const int tiles = M / 16;           // 3125

    // Workspace layout (~3.4 MB total):
    //   w1p  : 8*4*32*16  bf16 = 32 KB   (WMMA B layout)
    //   w2p  : 32*4*32*16 bf16 = 128 KB  (WMMA B layout)
    //   psum : 128*tiles f32
    //   psq  : 128*tiles f32
    //   scale/shift : 128 f32 each
    char* ws = (char*)d_ws;
    __bf16* w1p = (__bf16*)ws;
    __bf16* w2p = (__bf16*)(ws + 32768);
    size_t off = 32768 + 131072;
    float* psum = (float*)(ws + off);
    float* psq  = (float*)(ws + off + (size_t)C_DIM * tiles * sizeof(float));
    float* scale = (float*)(ws + off + 2 * (size_t)C_DIM * tiles * sizeof(float));
    float* shift = scale + C_DIM;

    {
        int tot1 = 8 * 4 * 32 * 16;
        pack_w_bf16<<<(tot1 + 255) / 256, 256, 0, stream>>>(w1, w1p, 128, 8);
        int tot2 = 32 * 4 * 32 * 16;
        pack_w_bf16<<<(tot2 + 255) / 256, 256, 0, stream>>>(w2, w2p, 512, 32);
    }
    gemm1_stats_kernel<<<tiles, 32, 0, stream>>>(s_feats, w1p, b1, psum, psq);
    bn_finalize_kernel<<<C_DIM, 256, 0, stream>>>(psum, psq, gamma1, beta1,
                                                  scale, shift, tiles, M);
    fused_kernel<<<tiles, 32, 0, stream>>>(s_feats, w1p, b1, scale, shift,
                                           w2p, b2, inds, out);
}
